// EncoderBlock_10350871183568
// MI455X (gfx1250) — compile-verified, matrix operand order checked by reading, not run on hardware
//
#include <hip/hip_runtime.h>
#include <hip/hip_bf16.h>

typedef __attribute__((ext_vector_type(16))) _Float16 v16h;
typedef __attribute__((ext_vector_type(8)))  _Float16 v8h;
typedef __attribute__((ext_vector_type(4)))  _Float16 v4h;
typedef __attribute__((ext_vector_type(8)))  float    v8f;
typedef __attribute__((ext_vector_type(4)))  int      v4i;

#define E_DIM   1024
#define H_DIM   16
#define D_DIM   64
#define FF_DIM  4096
#define L_SEQ   2048
#define NEG_VAL (-10000.0f)

#if __has_builtin(__builtin_amdgcn_global_load_async_to_lds_b128)
#define ASYNC_LDS 1
#endif

static __device__ __forceinline__ v8f vzero8() {
    v8f v;
#pragma unroll
    for (int i = 0; i < 8; ++i) v[i] = 0.0f;
    return v;
}

// WMMA 16-bit A/B fragment from an LDS row holding a 32-half K-window.
// Lane hi-half selects K-halves [hi*8, +8) (VGPRs 0..3) and [16+hi*8, +8).
static __device__ __forceinline__ v16h load_frag(const _Float16* p, int hi) {
    union { v16h v; v8h h[2]; } u;
    u.h[0] = *reinterpret_cast<const v8h*>(p + hi * 8);
    u.h[1] = *reinterpret_cast<const v8h*>(p + 16 + hi * 8);
    return u.v;
}

// ---- 16-lane xor butterflies on the VALU (v_permlane16) -------------------
#if __has_builtin(__builtin_amdgcn_permlane16)
static __device__ __forceinline__ float pl16(float v, unsigned s0, unsigned s1) {
    union { float f; unsigned u; } a, b;
    a.f = v;
    b.u = __builtin_amdgcn_permlane16(a.u, a.u, s0, s1, false, false);
    return b.f;
}
static __device__ __forceinline__ float rowmax16(float v) {
    v = fmaxf(v, pl16(v, 0x67452301u, 0xEFCDAB89u));  // xor 1
    v = fmaxf(v, pl16(v, 0x54761032u, 0xDCFE98BAu));  // xor 2
    v = fmaxf(v, pl16(v, 0x32107654u, 0xBA98FEDCu));  // xor 4
    v = fmaxf(v, pl16(v, 0xFEDCBA98u, 0x76543210u));  // xor 8
    return v;
}
static __device__ __forceinline__ float rowsum16(float v) {
    v += pl16(v, 0x67452301u, 0xEFCDAB89u);
    v += pl16(v, 0x54761032u, 0xDCFE98BAu);
    v += pl16(v, 0x32107654u, 0xBA98FEDCu);
    v += pl16(v, 0xFEDCBA98u, 0x76543210u);
    return v;
}
#else
static __device__ __forceinline__ float rowmax16(float v) {
    for (int mk = 1; mk < 16; mk <<= 1) v = fmaxf(v, __shfl_xor(v, mk, 32));
    return v;
}
static __device__ __forceinline__ float rowsum16(float v) {
    for (int mk = 1; mk < 16; mk <<= 1) v += __shfl_xor(v, mk, 32);
    return v;
}
#endif

#ifdef ASYNC_LDS
typedef __attribute__((address_space(1))) v4i*  gptr_v4i;
typedef __attribute__((address_space(3))) v4i*  lptr_v4i;
typedef __attribute__((address_space(1))) void* gptr_void;
typedef __attribute__((address_space(3))) void* lptr_void;

static __device__ __forceinline__ void async_cp16(const _Float16* g, _Float16* l) {
    __builtin_amdgcn_global_load_async_to_lds_b128(
        (gptr_v4i)(gptr_void)const_cast<_Float16*>(g),
        (lptr_v4i)(lptr_void)l, 0, 0);
}
static __device__ __forceinline__ void wait_async0() {
#if __has_builtin(__builtin_amdgcn_s_wait_asynccnt)
    __builtin_amdgcn_s_wait_asynccnt(0);
#else
    asm volatile("s_wait_asynccnt 0" ::: "memory");
#endif
}
#endif

// ---------------------------------------------------------------------------
// Tiled f16 WMMA GEMM:  C[M,N] = A[M,K] * Bt[N,K]^T  (Bt is B transposed)
// 256 threads (8 waves), block tile 128x128, wave tile 32x64, K-step 32,
// LDS double-buffered; tile t+1 streams (async if available) during tile t.
// ---------------------------------------------------------------------------
template <bool BIAS, bool RES, bool RELU, bool OUT16>
__global__ __launch_bounds__(256, 1)
void gemm_wmma_f16(const _Float16* __restrict__ A, const _Float16* __restrict__ Bt,
                   const float* __restrict__ bias, const float* __restrict__ res,
                   float* __restrict__ outF, _Float16* __restrict__ outH,
                   int M, int N, int K)
{
    __shared__ alignas(16) _Float16 As[2][128][40];   // 32-half rows, padded to 40
    __shared__ alignas(16) _Float16 Bs[2][128][40];

    const int tid  = threadIdx.x;
    const int lane = tid & 31;
    const int wave = tid >> 5;
    const int hi   = lane >> 4;
    const int ln15 = lane & 15;
    const int m0   = blockIdx.y * 128;
    const int n0   = blockIdx.x * 128;
    const int wr   = (wave >> 1) * 32;
    const int wc   = (wave & 1) * 64;

    v8f acc[2][4];
#pragma unroll
    for (int i = 0; i < 2; ++i)
#pragma unroll
        for (int j = 0; j < 4; ++j) acc[i][j] = vzero8();

    const int ldRow = tid >> 1;          // 0..127
    const int ldSeg = (tid & 1) * 16;    // 0 or 16 halfs
    const _Float16* aG = A  + (size_t)(m0 + ldRow) * K + ldSeg;
    const _Float16* bG = Bt + (size_t)(n0 + ldRow) * K + ldSeg;
    const int nk = K >> 5;

    auto compute = [&](int buf) {
        v16h af[2], bf[4];
        af[0] = load_frag(&As[buf][wr + ln15][0], hi);
        af[1] = load_frag(&As[buf][wr + 16 + ln15][0], hi);
#pragma unroll
        for (int j = 0; j < 4; ++j)
            bf[j] = load_frag(&Bs[buf][wc + j * 16 + ln15][0], hi);
#pragma unroll
        for (int i = 0; i < 2; ++i)
#pragma unroll
            for (int j = 0; j < 4; ++j)
                acc[i][j] = __builtin_amdgcn_wmma_f32_16x16x32_f16(
                    false, af[i], false, bf[j], (short)0, acc[i][j], false, false);
    };

#ifdef ASYNC_LDS
    // prologue: stream tile 0 into buffer 0 via the async-LDS path
    async_cp16(aG,      &As[0][ldRow][ldSeg]);
    async_cp16(aG + 8,  &As[0][ldRow][ldSeg + 8]);
    async_cp16(bG,      &Bs[0][ldRow][ldSeg]);
    async_cp16(bG + 8,  &Bs[0][ldRow][ldSeg + 8]);
    wait_async0();
    __syncthreads();
    for (int t = 0; t < nk; ++t) {
        const int cur = t & 1;
        if (t + 1 < nk) {
            const int kt = (t + 1) * 32;
            async_cp16(aG + kt,     &As[cur ^ 1][ldRow][ldSeg]);
            async_cp16(aG + kt + 8, &As[cur ^ 1][ldRow][ldSeg + 8]);
            async_cp16(bG + kt,     &Bs[cur ^ 1][ldRow][ldSeg]);
            async_cp16(bG + kt + 8, &Bs[cur ^ 1][ldRow][ldSeg + 8]);
            if (t + 2 < nk) {       // L2 prefetch two tiles ahead
                __builtin_prefetch(aG + kt + 32, 0, 1);
                __builtin_prefetch(bG + kt + 32, 0, 1);
            }
        }
        compute(cur);
        wait_async0();
        __syncthreads();
    }
#else
    {   // prologue (synchronous fallback): tile 0 -> buffer 0
        v8h a0 = *(const v8h*)(aG),     a1 = *(const v8h*)(aG + 8);
        v8h b0 = *(const v8h*)(bG),     b1 = *(const v8h*)(bG + 8);
        *(v8h*)&As[0][ldRow][ldSeg] = a0; *(v8h*)&As[0][ldRow][ldSeg + 8] = a1;
        *(v8h*)&Bs[0][ldRow][ldSeg] = b0; *(v8h*)&Bs[0][ldRow][ldSeg + 8] = b1;
    }
    __syncthreads();
    for (int t = 0; t < nk; ++t) {
        const int cur = t & 1;
        v8h a0, a1, b0, b1;
        if (t + 1 < nk) {            // issue next tile's global loads early
            const int kt = (t + 1) * 32;
            a0 = *(const v8h*)(aG + kt);     a1 = *(const v8h*)(aG + kt + 8);
            b0 = *(const v8h*)(bG + kt);     b1 = *(const v8h*)(bG + kt + 8);
            if (t + 2 < nk) {
                __builtin_prefetch(aG + kt + 32, 0, 1);
                __builtin_prefetch(bG + kt + 32, 0, 1);
            }
        }
        compute(cur);
        __syncthreads();
        if (t + 1 < nk) {
            *(v8h*)&As[cur ^ 1][ldRow][ldSeg]     = a0;
            *(v8h*)&As[cur ^ 1][ldRow][ldSeg + 8] = a1;
            *(v8h*)&Bs[cur ^ 1][ldRow][ldSeg]     = b0;
            *(v8h*)&Bs[cur ^ 1][ldRow][ldSeg + 8] = b1;
        }
        __syncthreads();
    }
#endif

    // C layout: lane holds column n = lane&15; VGPR r holds row r + 8*hi.
#pragma unroll
    for (int i = 0; i < 2; ++i)
#pragma unroll
        for (int j = 0; j < 4; ++j)
#pragma unroll
            for (int r = 0; r < 8; ++r) {
                int row = m0 + wr + i * 16 + hi * 8 + r;
                int col = n0 + wc + j * 16 + ln15;
                float v = acc[i][j][r];
                if (BIAS) v += bias[col];
                if (RES)  v += res[(size_t)row * N + col];
                if (RELU) v = fmaxf(v, 0.0f);
                if (OUT16) outH[(size_t)row * N + col] = (_Float16)v;
                else       outF[(size_t)row * N + col] = v;
            }
}

// ---------------------------------------------------------------------------
// Flash attention with WMMA. 128 threads (4 waves) per block; block handles
// 64 q-rows of one (batch, head); each wave owns 16 q-rows. Key chunk = 32.
// ---------------------------------------------------------------------------
__global__ __launch_bounds__(128, 1)
void attention_wmma(const _Float16* __restrict__ Q, const _Float16* __restrict__ K,
                    const _Float16* __restrict__ V, const int* __restrict__ mask,
                    _Float16* __restrict__ O)
{
    __shared__ alignas(16) _Float16 Qs[64][72];     // [q][d]
    __shared__ alignas(16) _Float16 Ks[32][72];     // [key][d] (B-layout for QK^T)
    __shared__ alignas(16) _Float16 Vt[64][40];     // [d][key] (B-layout for PV)
    __shared__ alignas(16) _Float16 Ps[4][16][40];  // per-wave P bounce [q][key]

    const int tid  = threadIdx.x;
    const int lane = tid & 31;
    const int wave = tid >> 5;
    const int hi   = lane >> 4;
    const int ln15 = lane & 15;
    const int q0   = blockIdx.x * 64;
    const int hh   = blockIdx.y;
    const int nb   = blockIdx.z;
    const size_t baseTok = (size_t)nb * L_SEQ;
    const int co   = hh * D_DIM;

    if (tid < 64) {  // load Q tile once
        const _Float16* src = Q + (baseTok + q0 + tid) * E_DIM + co;
#pragma unroll
        for (int j = 0; j < 8; ++j)
            *(v8h*)&Qs[tid][j * 8] = *(const v8h*)(src + j * 8);
    }

    v8f o[4];
#pragma unroll
    for (int j = 0; j < 4; ++j) o[j] = vzero8();
    float mr[8], lr[8];
#pragma unroll
    for (int r = 0; r < 8; ++r) { mr[r] = -3.0e38f; lr[r] = 0.0f; }

    const int krow = tid >> 2;           // 0..31
    const int kseg = (tid & 3) * 16;     // 0,16,32,48
    const int* mrow = mask + nb * L_SEQ;
    _Float16 (*Pw)[40] = Ps[wave];

    for (int key0 = 0; key0 < L_SEQ; key0 += 32) {
        __syncthreads();
        {   // stage K rows (no transpose needed) and V transposed
            const _Float16* ksrc = K + (baseTok + key0 + krow) * E_DIM + co + kseg;
            *(v8h*)&Ks[krow][kseg]     = *(const v8h*)(ksrc);
            *(v8h*)&Ks[krow][kseg + 8] = *(const v8h*)(ksrc + 8);
            const _Float16* vsrc = V + (baseTok + key0 + krow) * E_DIM + co + kseg;
            v8h v0 = *(const v8h*)(vsrc);
            v8h v1 = *(const v8h*)(vsrc + 8);
#pragma unroll
            for (int j = 0; j < 8; ++j) Vt[kseg + j][krow]     = v0[j];
#pragma unroll
            for (int j = 0; j < 8; ++j) Vt[kseg + 8 + j][krow] = v1[j];
        }
        __syncthreads();

        // S = Q(16x64) @ K^T(64x32): 2 key-subtiles x 2 k-steps
        v8f s[2];
        s[0] = vzero8(); s[1] = vzero8();
#pragma unroll
        for (int ks = 0; ks < 2; ++ks) {
            v16h qf = load_frag(&Qs[wave * 16 + ln15][ks * 32], hi);
#pragma unroll
            for (int c = 0; c < 2; ++c) {
                v16h kf = load_frag(&Ks[c * 16 + ln15][ks * 32], hi);
                s[c] = __builtin_amdgcn_wmma_f32_16x16x32_f16(
                    false, qf, false, kf, (short)0, s[c], false, false);
            }
        }

        // mask (before scaling, matching reference) + online softmax
        const int mv0 = mrow[key0 + ln15];
        const int mv1 = mrow[key0 + 16 + ln15];
#pragma unroll
        for (int r = 0; r < 8; ++r) {
            float a = s[0][r]; if (mv0 == 0) a = NEG_VAL; a *= 0.125f;
            float b = s[1][r]; if (mv1 == 0) b = NEG_VAL; b *= 0.125f;
            float rm   = rowmax16(fmaxf(a, b));
            float mn   = fmaxf(mr[r], rm);
            float corr = __expf(mr[r] - mn);
            mr[r] = mn;
            float p0 = __expf(a - mn);
            float p1 = __expf(b - mn);
            float rs = rowsum16(p0 + p1);
            lr[r] = lr[r] * corr + rs;
#pragma unroll
            for (int j = 0; j < 4; ++j) o[j][r] *= corr;
            Pw[hi * 8 + r][ln15]      = (_Float16)p0;   // C-layout -> LDS
            Pw[hi * 8 + r][16 + ln15] = (_Float16)p1;
        }
        asm volatile("s_wait_dscnt 0" ::: "memory");    // within-wave LDS RAW

        // O += P(16x32) @ V(32x64)
        v16h pf = load_frag(&Pw[ln15][0], hi);
#pragma unroll
        for (int j = 0; j < 4; ++j) {
            v16h vf = load_frag(&Vt[j * 16 + ln15][0], hi);
            o[j] = __builtin_amdgcn_wmma_f32_16x16x32_f16(
                false, pf, false, vf, (short)0, o[j], false, false);
        }
    }

#pragma unroll
    for (int r = 0; r < 8; ++r) {
        float inv = 1.0f / lr[r];
        int q = q0 + wave * 16 + hi * 8 + r;
        _Float16* dst = O + (baseTok + q) * E_DIM + co;
#pragma unroll
        for (int j = 0; j < 4; ++j)
            dst[j * 16 + ln15] = (_Float16)(o[j][r] * inv);
    }
}

// ---------------------------------------------------------------------------
// LayerNorm over rows of E_DIM floats; writes f32 and optionally f16 copy.
// ---------------------------------------------------------------------------
__global__ __launch_bounds__(256, 1)
void layernorm_k(const float* __restrict__ in, const float* __restrict__ g,
                 const float* __restrict__ b, float* __restrict__ outF,
                 _Float16* __restrict__ outH)
{
    const int row = blockIdx.x;
    const int t   = threadIdx.x;
    const float4 x = ((const float4*)(in + (size_t)row * E_DIM))[t];
    float s  = x.x + x.y + x.z + x.w;
    float sq = x.x * x.x + x.y * x.y + x.z * x.z + x.w * x.w;
#pragma unroll
    for (int mk = 1; mk < 32; mk <<= 1) { s += __shfl_xor(s, mk, 32); sq += __shfl_xor(sq, mk, 32); }
    __shared__ float sh_s[8], sh_q[8];
    const int wave = t >> 5;
    if ((t & 31) == 0) { sh_s[wave] = s; sh_q[wave] = sq; }
    __syncthreads();
    float ts = 0.0f, tq = 0.0f;
#pragma unroll
    for (int i = 0; i < 8; ++i) { ts += sh_s[i]; tq += sh_q[i]; }
    const float mean = ts * (1.0f / E_DIM);
    const float var  = tq * (1.0f / E_DIM) - mean * mean;
    const float inv  = rsqrtf(var + 1e-6f);
    const float4 gg = ((const float4*)g)[t];
    const float4 bb = ((const float4*)b)[t];
    float4 y;
    y.x = gg.x * (x.x - mean) * inv + bb.x;
    y.y = gg.y * (x.y - mean) * inv + bb.y;
    y.z = gg.z * (x.z - mean) * inv + bb.z;
    y.w = gg.w * (x.w - mean) * inv + bb.w;
    ((float4*)(outF + (size_t)row * E_DIM))[t] = y;
    if (outH) {
        v4h h; h[0] = (_Float16)y.x; h[1] = (_Float16)y.y;
               h[2] = (_Float16)y.z; h[3] = (_Float16)y.w;
        *(v4h*)(outH + (size_t)row * E_DIM + t * 4) = h;
    }
}

__global__ void cvt_f32_to_f16(const float* __restrict__ in, _Float16* __restrict__ out, int n4)
{
    int i = blockIdx.x * blockDim.x + threadIdx.x;
    if (i < n4) {
        float4 v = ((const float4*)in)[i];
        v4h h; h[0] = (_Float16)v.x; h[1] = (_Float16)v.y;
               h[2] = (_Float16)v.z; h[3] = (_Float16)v.w;
        ((v4h*)out)[i] = h;
    }
}

// in[K][N] (f32) -> out[N][K] (f16), LDS-tiled for coalescing both ways
__global__ __launch_bounds__(256, 1)
void cvt_transpose_f16(const float* __restrict__ in, _Float16* __restrict__ out, int K, int N)
{
    __shared__ float tile[32][33];
    int n0 = blockIdx.x * 32, k0 = blockIdx.y * 32;
    int tx = threadIdx.x, ty = threadIdx.y;       // (32, 8)
#pragma unroll
    for (int i = 0; i < 4; ++i)
        tile[ty + i * 8][tx] = in[(size_t)(k0 + ty + i * 8) * N + n0 + tx];
    __syncthreads();
#pragma unroll
    for (int i = 0; i < 4; ++i)
        out[(size_t)(n0 + ty + i * 8) * K + k0 + tx] = (_Float16)tile[tx][ty + i * 8];
}

extern "C" void kernel_launch(void* const* d_in, const int* in_sizes, int n_in,
                              void* d_out, int out_size, void* d_ws, size_t ws_size,
                              hipStream_t stream)
{
    (void)in_sizes; (void)n_in; (void)out_size; (void)ws_size;
    const float* x   = (const float*)d_in[0];
    const int*   msk = (const int*)d_in[1];
    const float* wq  = (const float*)d_in[2];
    const float* wk  = (const float*)d_in[3];
    const float* wv  = (const float*)d_in[4];
    const float* wo  = (const float*)d_in[5];
    const float* bo  = (const float*)d_in[6];
    const float* g1  = (const float*)d_in[7];
    const float* be1 = (const float*)d_in[8];
    const float* g2  = (const float*)d_in[9];
    const float* be2 = (const float*)d_in[10];
    const float* w1  = (const float*)d_in[11];
    const float* b1  = (const float*)d_in[12];
    const float* w2  = (const float*)d_in[13];
    const float* b2  = (const float*)d_in[14];

    const int NL = 2 * L_SEQ;                       // 4096 tokens
    char* ws = (char*)d_ws;
    const size_t MB = 1024ull * 1024ull;
    _Float16* AH    = (_Float16*)(ws + 0);          //  8 MB: x f16, later h f16
    _Float16* FFH   = (_Float16*)(ws + 8 * MB);     // 32 MB: relu(h@w1+b1) f16
    _Float16* WT    = (_Float16*)(ws + 40 * MB);    //  8 MB: current weight^T f16
    _Float16* QH    = (_Float16*)(ws + 48 * MB);    //  8 MB
    _Float16* KH    = (_Float16*)(ws + 56 * MB);    //  8 MB
    _Float16* VH    = (_Float16*)(ws + 64 * MB);    //  8 MB
    _Float16* ATTNH = (_Float16*)(ws + 72 * MB);    //  8 MB
    float*    RES   = (float*)(ws + 80 * MB);       // 16 MB (pre-LN1, later pre-LN2)
    float*    HBUF  = (float*)(ws + 96 * MB);       // 16 MB: h f32 (residual)

    dim3 blk256(256), blk128(128), tblk(32, 8);
    dim3 gP(E_DIM / 128, NL / 128);                 // (8, 32)
    dim3 gF1(FF_DIM / 128, NL / 128);               // (32, 32)

    cvt_f32_to_f16<<<(NL * E_DIM / 4 + 255) / 256, blk256, 0, stream>>>(x, AH, NL * E_DIM / 4);

    // Q / K / V projections (weight buffer reused; stream order serializes)
    cvt_transpose_f16<<<dim3(E_DIM / 32, E_DIM / 32), tblk, 0, stream>>>(wq, WT, E_DIM, E_DIM);
    gemm_wmma_f16<false, false, false, true><<<gP, blk256, 0, stream>>>(AH, WT, nullptr, nullptr, nullptr, QH, NL, E_DIM, E_DIM);
    cvt_transpose_f16<<<dim3(E_DIM / 32, E_DIM / 32), tblk, 0, stream>>>(wk, WT, E_DIM, E_DIM);
    gemm_wmma_f16<false, false, false, true><<<gP, blk256, 0, stream>>>(AH, WT, nullptr, nullptr, nullptr, KH, NL, E_DIM, E_DIM);
    cvt_transpose_f16<<<dim3(E_DIM / 32, E_DIM / 32), tblk, 0, stream>>>(wv, WT, E_DIM, E_DIM);
    gemm_wmma_f16<false, false, false, true><<<gP, blk256, 0, stream>>>(AH, WT, nullptr, nullptr, nullptr, VH, NL, E_DIM, E_DIM);

    attention_wmma<<<dim3(L_SEQ / 64, H_DIM, 2), blk128, 0, stream>>>(QH, KH, VH, msk, ATTNH);

    // out-proj + bo + residual x  -> RES (f32)
    cvt_transpose_f16<<<dim3(E_DIM / 32, E_DIM / 32), tblk, 0, stream>>>(wo, WT, E_DIM, E_DIM);
    gemm_wmma_f16<true, true, false, false><<<gP, blk256, 0, stream>>>(ATTNH, WT, bo, x, RES, nullptr, NL, E_DIM, E_DIM);

    // LN1 -> h (f32 in HBUF, f16 in AH)
    layernorm_k<<<NL, blk256, 0, stream>>>(RES, g1, be1, HBUF, AH);

    // FF1: relu(h @ w1 + b1) -> FFH (f16)
    cvt_transpose_f16<<<dim3(FF_DIM / 32, E_DIM / 32), tblk, 0, stream>>>(w1, WT, E_DIM, FF_DIM);
    gemm_wmma_f16<true, false, true, true><<<gF1, blk256, 0, stream>>>(AH, WT, b1, nullptr, nullptr, FFH, NL, FF_DIM, E_DIM);

    // FF2: ff @ w2 + b2 + h -> RES (f32)
    cvt_transpose_f16<<<dim3(E_DIM / 32, FF_DIM / 32), tblk, 0, stream>>>(w2, WT, FF_DIM, E_DIM);
    gemm_wmma_f16<true, true, false, false><<<gP, blk256, 0, stream>>>(FFH, WT, b2, HBUF, RES, nullptr, NL, E_DIM, FF_DIM);

    // LN2 -> final output
    layernorm_k<<<NL, blk256, 0, stream>>>(RES, g2, be2, (float*)d_out, nullptr);
}
